// InteractiveBidirectionalMambaGRU_20701742366782
// MI455X (gfx1250) — compile-verified, matrix operand order checked by reading, not run on hardware
//
#include <hip/hip_runtime.h>
#include <hip/hip_bf16.h>

// ---------------------------------------------------------------------------
// InteractiveBidirectionalMambaGRU for MI455X (gfx1250, wave32, WMMA + TDM)
// GEMMs: v_wmma_f32_16x16x32_f16 (f16 in, f32 acc). Tile staging via the
// Tensor Data Mover, double-buffered so DMA overlaps WMMA:
//   issue tile t+1 -> s_wait_tensorcnt 2 -> barrier -> compute tile t
// Falls back to synchronous global->LDS staging if the builtin is absent.
// ---------------------------------------------------------------------------

typedef __attribute__((ext_vector_type(16))) _Float16 v16h;
typedef __attribute__((ext_vector_type(8)))  _Float16 v8h;
typedef __attribute__((ext_vector_type(8)))  float    v8f;

#define D_MODEL 256
#define D_STATE 16
#define DT_RANK 16
#define D_CONV  4
#define BATCH   2
#define NPATCH  32
#define NTOK    128
#define BN      256      // BATCH*NTOK sequences
#define MROWS   8192     // BN*NPATCH activation rows
#define DBC_LD  64       // padded leading dim of dbc (48 -> 64, guard-free GEMM)

union HV { v16h v; v8h h[2]; };

#if defined(__AMDGCN__) && __has_builtin(__builtin_amdgcn_tensor_load_to_lds) && \
    __has_builtin(__builtin_amdgcn_s_wait_tensorcnt)
#define USE_TDM 1
#else
#define USE_TDM 0
#endif

#if USE_TDM
typedef __attribute__((ext_vector_type(4))) unsigned int u32x4;
typedef __attribute__((ext_vector_type(8))) int          i32x8;
typedef __attribute__((ext_vector_type(4))) int          i32x4;

// Issue one TDM descriptor moving a 64x32 f16 tile (row stride = strideElems
// f16 elements) from global memory to LDS byte offset ldsOff.
// D# layout per CDNA5 ISA (2D tensor, 2 groups).
__device__ __forceinline__ void tdm_load_tile_64x32(unsigned ldsOff,
                                                    unsigned long long gaddr,
                                                    int strideElems) {
  u32x4 g0;
  g0[0] = 1u;                                                    // count=1
  g0[1] = ldsOff;                                                // lds_addr
  g0[2] = (unsigned)gaddr;                                       // gaddr[31:0]
  g0[3] = (unsigned)((gaddr >> 32) & 0x01FFFFFFull) | (2u << 30);// [56:32]|type=2
  i32x8 g1;
  g1[0] = 1 << 16;               // data_size = 2 bytes
  g1[1] = (int)(32u << 16);      // tensor_dim0 = 32 (bits 63:48 of group1)
  g1[2] = (int)(64u << 16);      // tensor_dim0 hi = 0 | tensor_dim1 = 64
  g1[3] = (int)(32u << 16);      // tensor_dim1 hi = 0 | tile_dim0 = 32
  g1[4] = 64;                    // tile_dim1 = 64, tile_dim2 = 0
  g1[5] = strideElems;           // tensor_dim0_stride[31:0]
  g1[6] = 0;                     // stride hi / dim1_stride lo
  g1[7] = 0;
  i32x4 z4 = {0, 0, 0, 0};
#if __clang_major__ >= 23
  i32x8 z8 = {0, 0, 0, 0, 0, 0, 0, 0};
  __builtin_amdgcn_tensor_load_to_lds(g0, g1, z4, z4, z8, 0);
#else
  __builtin_amdgcn_tensor_load_to_lds(g0, g1, z4, z4, 0);
#endif
}
#endif  // USE_TDM

// ---------------------------------------------------------------------------
// small packing / conversion kernels
// ---------------------------------------------------------------------------
__global__ void cvt_f32_f16(const float* __restrict__ src, _Float16* __restrict__ dst, int n) {
  int i = blockIdx.x * 256 + threadIdx.x;
  if (i < n) dst[i] = (_Float16)src[i];
}

// X16[row=bn*32+p][d] = x[bi, p, ni, d]   (bn = bi*128+ni)
__global__ void pack_x_f16(const float* __restrict__ x, _Float16* __restrict__ X16) {
  int gid = blockIdx.x * 256 + threadIdx.x;
  if (gid >= MROWS * D_MODEL) return;
  int d   = gid & 255;
  int row = gid >> 8;
  int p   = row & 31, bn = row >> 5;
  int ni  = bn & 127, bi = bn >> 7;
  size_t src = (((size_t)bi * NPATCH + p) * NTOK + ni) * D_MODEL + d;
  X16[gid] = (_Float16)x[src];
}

// xproj_w (3,48,256) -> f16 zero-padded (3,64,256)
__global__ void pack_xpw(const float* __restrict__ xproj_w, _Float16* __restrict__ dst) {
  int gid = blockIdx.x * 256 + threadIdx.x;
  if (gid >= 3 * 64 * 256) return;
  int k = gid & 255;
  int r = (gid >> 8) & 63;
  int m = gid >> 14;
  dst[gid] = (r < 48) ? (_Float16)xproj_w[((size_t)m * 48 + r) * 256 + k]
                      : (_Float16)0.f;
}

// dt_w (3,256,16) -> f16 padded (3,256,32)
__global__ void pack_dtw(const float* __restrict__ dt_w, _Float16* __restrict__ dst) {
  int gid = blockIdx.x * 256 + threadIdx.x;
  if (gid >= 3 * 256 * 32) return;
  int k = gid & 31, md = gid >> 5;
  dst[gid] = (k < DT_RANK) ? (_Float16)dt_w[md * DT_RANK + k] : (_Float16)0.f;
}

// dt_r = dbc[..., 0:16] -> f16 padded (3, 8192, 32)
__global__ void pack_dtr(const float* __restrict__ dbc, _Float16* __restrict__ dst) {
  int gid = blockIdx.x * 256 + threadIdx.x;
  if (gid >= 3 * MROWS * 32) return;
  int k = gid & 31, r = gid >> 5;
  dst[gid] = (k < DT_RANK) ? (_Float16)dbc[(size_t)r * DBC_LD + k] : (_Float16)0.f;
}

// ---------------------------------------------------------------------------
// generic WMMA GEMM:  C[M,N] = A[M,K] * B[N,K]^T   (A,B f16 row-major, C f32)
// block = 128 threads = 4 waves; block tile 64x64; K-step 32.
// Requires: M,N multiples of 64; K multiple of 32 (all call sites satisfy).
// LDS double buffer: [A0|B0|A1|B1] at byte offsets 0/4096/8192/12288.
// ---------------------------------------------------------------------------
__global__ __launch_bounds__(128)
void gemm_f16_wmma(const _Float16* __restrict__ A, long aStride,
                   const _Float16* __restrict__ B, long bStride,
                   float* __restrict__ C, long cStride,
                   int M, int N, int K) {
  __shared__ _Float16 smem[4 * 64 * 32];   // 16 KB, LDS offset 0 by construction

  const _Float16* Ab = A + (size_t)blockIdx.z * aStride;
  const _Float16* Bb = B + (size_t)blockIdx.z * bStride;
  float*          Cb = C + (size_t)blockIdx.z * cStride;
  const int m0 = blockIdx.y * 64;
  const int n0 = blockIdx.x * 64;
  const int tid  = threadIdx.x;
  const int lane = tid & 31;
  const int wave = tid >> 5;
  const int rsel = lane & 15;          // M (A) / N (B) selector within 16
  const int kb   = (lane >> 4) * 8;    // per-ISA K base for this lane half
  const int nk   = K >> 5;

  v8f acc0 = {}, acc1 = {}, acc2 = {}, acc3 = {};

#if USE_TDM
  // Opaque may-write so the compiler cannot prove smem is never stored (the
  // TDM builtin writes LDS behind the alias analysis' back). Never true.
  if (M == -1) smem[tid] = (_Float16)0.f;
  // prologue: wave 0 issues tile 0 into buffer 0
  if (tid < 32) {
    tdm_load_tile_64x32(0u,
        (unsigned long long)(size_t)(Ab + (size_t)m0 * K), K);
    tdm_load_tile_64x32(4096u,
        (unsigned long long)(size_t)(Bb + (size_t)n0 * K), K);
  }
#endif

  for (int t = 0; t < nk; ++t) {
    const int cur = t & 1;
    _Float16* As = smem + cur * 4096;        // elements; 8192B per buffer pair
    _Float16* Bs = As + 2048;

#if USE_TDM
    // issue tile t+1 into the other buffer, then drain tile t's pair only
    if (t + 1 < nk) {
      if (tid < 32) {
        int kk = (t + 1) * 32;
        tdm_load_tile_64x32((unsigned)((cur ^ 1) * 8192),
            (unsigned long long)(size_t)(Ab + (size_t)m0 * K + kk), K);
        tdm_load_tile_64x32((unsigned)((cur ^ 1) * 8192 + 4096),
            (unsigned long long)(size_t)(Bb + (size_t)n0 * K + kk), K);
      }
      __builtin_amdgcn_s_wait_tensorcnt(2);
    } else {
      __builtin_amdgcn_s_wait_tensorcnt(0);
    }
    asm volatile("" ::: "memory");           // LDS now holds tile t
    __syncthreads();
#else
    // fallback: synchronous staging, 128 thr * 2 v8h each per matrix
    #pragma unroll
    for (int u = 0; u < 2; ++u) {
      int idx = tid * 2 + u;
      int r = idx >> 2;
      int c = (idx & 3) * 8;
      *(v8h*)&As[r * 32 + c] = *(const v8h*)&Ab[(size_t)(m0 + r) * K + t * 32 + c];
      *(v8h*)&Bs[r * 32 + c] = *(const v8h*)&Bb[(size_t)(n0 + r) * K + t * 32 + c];
    }
    __syncthreads();
    if (t + 1 < nk)  // global_prefetch_b8 on gfx1250
      __builtin_prefetch(&Ab[(size_t)(m0 + (tid >> 1)) * K + (t + 1) * 32], 0, 1);
#endif

    // ---- fragments per CDNA5 16-bit 16x32 layout: lane holds row rsel,
    //      K in [kb,kb+8) and [kb+16,kb+24)
    HV fa, fb;
    fa.h[0] = *(const v8h*)&As[(wave * 16 + rsel) * 32 + kb];
    fa.h[1] = *(const v8h*)&As[(wave * 16 + rsel) * 32 + kb + 16];
    #pragma unroll
    for (int j = 0; j < 4; ++j) {
      fb.h[0] = *(const v8h*)&Bs[(j * 16 + rsel) * 32 + kb];
      fb.h[1] = *(const v8h*)&Bs[(j * 16 + rsel) * 32 + kb + 16];
      v8f* accp = (j == 0) ? &acc0 : (j == 1) ? &acc1 : (j == 2) ? &acc2 : &acc3;
      *accp = __builtin_amdgcn_wmma_f32_16x16x32_f16(
          false, fa.v, false, fb.v, (short)0, *accp, false, false);
    }
    // all waves done reading tile t before its buffer can be re-targeted
    __syncthreads();
  }

  // ---- store per C layout: VGPR v -> row (v + (lane>>4)*8), lane -> col
  const int rbase = m0 + wave * 16 + (lane >> 4) * 8;
  #pragma unroll
  for (int j = 0; j < 4; ++j) {
    const v8f acc = (j == 0) ? acc0 : (j == 1) ? acc1 : (j == 2) ? acc2 : acc3;
    int col = n0 + j * 16 + rsel;
    #pragma unroll
    for (int v = 0; v < 8; ++v)
      Cb[(size_t)(rbase + v) * N + col] = acc[v];
  }
}

// ---------------------------------------------------------------------------
// causal depthwise conv (D_CONV=4) along p + bias + SiLU
// ---------------------------------------------------------------------------
__global__ void conv_silu(const float* __restrict__ xz,
                          const float* __restrict__ conv_w,
                          const float* __restrict__ conv_b,
                          float* __restrict__ xc32,
                          _Float16* __restrict__ xc16) {
  int gid = blockIdx.x * 256 + threadIdx.x;
  if (gid >= 3 * MROWS * D_MODEL) return;
  int d   = gid & 255;
  int row = (gid >> 8) & (MROWS - 1);
  int m   = gid >> 21;
  int p   = row & 31;
  float acc = conv_b[m * 256 + d];
  #pragma unroll
  for (int j = 0; j < D_CONV; ++j) {
    int pp = p + j - (D_CONV - 1);
    if (pp >= 0) {
      size_t srow = (size_t)m * MROWS + (row - p + pp);
      acc += xz[srow * 512 + d] * conv_w[(m * 256 + d) * D_CONV + j];
    }
  }
  float s = acc / (1.f + expf(-acc));  // SiLU
  xc32[gid] = s;
  xc16[gid] = (_Float16)s;
}

// ---------------------------------------------------------------------------
// selective scan: per (m, bn) block, thread d owns 16 states in registers
// emits y_gated = (scan_y + xc*D_skip) * silu(zg) as f16 for out_proj
// ---------------------------------------------------------------------------
__global__ __launch_bounds__(256)
void ssm_scan(const float* __restrict__ dbc,  const float* __restrict__ dtlin,
              const float* __restrict__ dt_b, const float* __restrict__ A_log,
              const float* __restrict__ D_skip, const float* __restrict__ xc32,
              const float* __restrict__ xz,   _Float16* __restrict__ yg16) {
  int m  = blockIdx.x / BN;
  int bn = blockIdx.x % BN;
  int d  = threadIdx.x;
  __shared__ float Bt[D_STATE], Ct[D_STATE];
  float Areg[D_STATE], h[D_STATE];
  #pragma unroll
  for (int s = 0; s < D_STATE; ++s) {
    Areg[s] = -expf(A_log[((size_t)m * 256 + d) * D_STATE + s]);
    h[s] = 0.f;
  }
  float dskip = D_skip[m * 256 + d];
  float dtb   = dt_b[m * 256 + d];
  for (int p = 0; p < NPATCH; ++p) {
    int row = bn * 32 + p;
    size_t b64 = ((size_t)m * MROWS + row) * DBC_LD;
    __syncthreads();
    if (d < 16)       Bt[d]      = dbc[b64 + DT_RANK + d];
    else if (d < 32)  Ct[d - 16] = dbc[b64 + DT_RANK + D_STATE + (d - 16)];
    __syncthreads();
    size_t idx = ((size_t)m * MROWS + row) * 256 + d;
    float dl  = dtlin[idx] + dtb;
    float dt  = (dl > 20.f) ? dl : log1pf(expf(dl));  // softplus
    float xcv = xc32[idx];
    float dtx = dt * xcv;
    float y = 0.f;
    #pragma unroll
    for (int s = 0; s < D_STATE; ++s) {
      float dA = expf(dt * Areg[s]);
      h[s] = dA * h[s] + dtx * Bt[s];
      y += h[s] * Ct[s];
    }
    y += xcv * dskip;
    float zg = xz[((size_t)m * MROWS + row) * 512 + 256 + d];
    float sg = zg / (1.f + expf(-zg));
    yg16[idx] = (_Float16)(y * sg);
  }
}

// z = sigmoid(outs[0]), r = sigmoid(outs[1]), hc = tanh(outs[2])  (in place;
// outs is fully rewritten by the preceding GEMM every call)
__global__ void act_zrh(float* __restrict__ outs) {
  int gid = blockIdx.x * 256 + threadIdx.x;
  if (gid >= 3 * MROWS * D_MODEL) return;
  int m = gid >> 21;
  float v = outs[gid];
  outs[gid] = (m < 2) ? 1.f / (1.f + expf(-v)) : tanhf(v);
}

// GRU input concat pack for step p (p>=1): dir0=[hf[p-1],hb[p]], dir1=[hf[bp],hb[bp+1]]
__global__ void pack_gru(const float* __restrict__ hf, const float* __restrict__ hb,
                         _Float16* __restrict__ Acat, int p) {
  int gid = blockIdx.x * 256 + threadIdx.x;
  if (gid >= 2 * BN * 512) return;
  int k   = gid & 511;
  int bn  = (gid >> 9) & 255;
  int dir = gid >> 17;
  float v;
  if (dir == 0) {
    v = (k < 256) ? hf[((size_t)bn * 32 + (p - 1)) * 256 + k]
                  : hb[((size_t)bn * 32 + p) * 256 + (k - 256)];
  } else {
    int bp = NPATCH - 1 - p;
    v = (k < 256) ? hf[((size_t)bn * 32 + bp) * 256 + k]
                  : hb[((size_t)bn * 32 + (bp + 1)) * 256 + (k - 256)];
  }
  Acat[gid] = (_Float16)v;
}

__global__ void gru_update(const float* __restrict__ gtmp, const float* __restrict__ outs,
                           const float* __restrict__ bf, const float* __restrict__ bb,
                           float* __restrict__ hf, float* __restrict__ hb,
                           int p, int haveInput) {
  int gid = blockIdx.x * 256 + threadIdx.x;
  if (gid >= 2 * BN * D_MODEL) return;
  int d   = gid & 255;
  int bn  = (gid >> 8) & 255;
  int dir = gid >> 16;
  int q   = dir ? (NPATCH - 1 - p) : p;
  size_t oi = ((size_t)bn * 32 + q) * 256 + d;
  float hi = 0.f;
  if (haveInput)
    hi = gtmp[(size_t)dir * BN * 256 + (size_t)bn * 256 + d] + (dir ? bb[d] : bf[d]);
  const float* z  = outs;
  const float* r  = outs + (size_t)MROWS * 256;
  const float* hc = outs + (size_t)2 * MROWS * 256;
  float zv = z[oi], rv = r[oi], hcv = hc[oi];
  float hn = tanhf(hcv + rv * hi);
  float nv = (1.f - zv) * hi + zv * hn;
  if (dir == 0) hf[oi] = nv; else hb[oi] = nv;
}

// combined = hf[:, P-1] + hb[:, 0]; LayerNorm over D
__global__ __launch_bounds__(256)
void final_ln(const float* __restrict__ hf, const float* __restrict__ hb,
              const float* __restrict__ ln_g, const float* __restrict__ ln_b,
              float* __restrict__ out) {
  int bn = blockIdx.x, d = threadIdx.x;
  float c = hf[((size_t)bn * 32 + (NPATCH - 1)) * 256 + d]
          + hb[((size_t)bn * 32 + 0) * 256 + d];
  __shared__ float red[256];
  red[d] = c; __syncthreads();
  for (int s = 128; s > 0; s >>= 1) { if (d < s) red[d] += red[d + s]; __syncthreads(); }
  float mu = red[0] * (1.f / 256.f);
  __syncthreads();
  float diff = c - mu;
  red[d] = diff * diff; __syncthreads();
  for (int s = 128; s > 0; s >>= 1) { if (d < s) red[d] += red[d + s]; __syncthreads(); }
  float var = red[0] * (1.f / 256.f);
  out[(size_t)bn * 256 + d] = diff * rsqrtf(var + 1e-5f) * ln_g[d] + ln_b[d];
}

// ---------------------------------------------------------------------------
extern "C" void kernel_launch(void* const* d_in, const int* in_sizes, int n_in,
                              void* d_out, int out_size, void* d_ws, size_t ws_size,
                              hipStream_t stream) {
  const float* x       = (const float*)d_in[0];
  const float* in_w    = (const float*)d_in[1];
  const float* conv_w  = (const float*)d_in[2];
  const float* conv_b  = (const float*)d_in[3];
  const float* xproj_w = (const float*)d_in[4];
  const float* dt_w    = (const float*)d_in[5];
  const float* dt_b    = (const float*)d_in[6];
  const float* A_log   = (const float*)d_in[7];
  const float* D_skip  = (const float*)d_in[8];
  const float* out_w   = (const float*)d_in[9];
  const float* wf      = (const float*)d_in[10];
  const float* bf      = (const float*)d_in[11];
  const float* wb      = (const float*)d_in[12];
  const float* bb      = (const float*)d_in[13];
  const float* ln_g    = (const float*)d_in[14];
  const float* ln_b    = (const float*)d_in[15];
  float* out = (float*)d_out;

  // ---- workspace carve-up (256B aligned)
  char* w = (char*)d_ws;
  size_t off = 0;
  auto alloc = [&](size_t bytes) -> void* {
    off = (off + 255) & ~(size_t)255;
    void* p = w + off;
    off += bytes;
    return p;
  };
  _Float16* X16     = (_Float16*)alloc(sizeof(_Float16) * MROWS * 256);
  _Float16* inw16   = (_Float16*)alloc(sizeof(_Float16) * 3 * 512 * 256);
  _Float16* xpw16   = (_Float16*)alloc(sizeof(_Float16) * 3 * 64 * 256);
  _Float16* dtw16   = (_Float16*)alloc(sizeof(_Float16) * 3 * 256 * 32);
  _Float16* outw16  = (_Float16*)alloc(sizeof(_Float16) * 3 * 256 * 256);
  _Float16* wcat16  = (_Float16*)alloc(sizeof(_Float16) * 2 * 256 * 512);
  _Float16* xc16    = (_Float16*)alloc(sizeof(_Float16) * 3 * MROWS * 256);
  _Float16* dtr16   = (_Float16*)alloc(sizeof(_Float16) * 3 * MROWS * 32);
  _Float16* yg16    = (_Float16*)alloc(sizeof(_Float16) * 3 * MROWS * 256);
  _Float16* Acat16  = (_Float16*)alloc(sizeof(_Float16) * 2 * 256 * 512);
  float* xz    = (float*)alloc(sizeof(float) * 3 * MROWS * 512);
  float* xc32  = (float*)alloc(sizeof(float) * 3 * MROWS * 256);
  float* dbc   = (float*)alloc(sizeof(float) * 3 * MROWS * DBC_LD);
  float* dtlin = (float*)alloc(sizeof(float) * 3 * MROWS * 256);
  float* outs  = (float*)alloc(sizeof(float) * 3 * MROWS * 256);
  float* gtmp  = (float*)alloc(sizeof(float) * 2 * 256 * 256);
  float* hf    = (float*)alloc(sizeof(float) * BN * 32 * 256);
  float* hb    = (float*)alloc(sizeof(float) * BN * 32 * 256);

  auto grid1 = [](int n) { return dim3((n + 255) / 256); };

  // ---- weight conversion / padding
  cvt_f32_f16<<<grid1(3 * 512 * 256), 256, 0, stream>>>(in_w, inw16, 3 * 512 * 256);
  pack_xpw<<<grid1(3 * 64 * 256), 256, 0, stream>>>(xproj_w, xpw16);
  cvt_f32_f16<<<grid1(3 * 256 * 256), 256, 0, stream>>>(out_w, outw16, 3 * 256 * 256);
  cvt_f32_f16<<<grid1(256 * 512), 256, 0, stream>>>(wf, wcat16, 256 * 512);
  cvt_f32_f16<<<grid1(256 * 512), 256, 0, stream>>>(wb, wcat16 + 256 * 512, 256 * 512);
  pack_dtw<<<grid1(3 * 256 * 32), 256, 0, stream>>>(dt_w, dtw16);

  // ---- activations
  pack_x_f16<<<grid1(MROWS * 256), 256, 0, stream>>>(x, X16);

  // in_proj: xz[m] = X @ in_w[m]^T   (M=8192, N=512, K=256, batch=3, shared A)
  gemm_f16_wmma<<<dim3(512 / 64, MROWS / 64, 3), 128, 0, stream>>>(
      X16, 0L, inw16, (long)512 * 256, xz, (long)MROWS * 512, MROWS, 512, 256);

  conv_silu<<<grid1(3 * MROWS * 256), 256, 0, stream>>>(xz, conv_w, conv_b, xc32, xc16);

  // xproj: dbc[m] = xc @ xproj_w[m]^T   (N padded to 64, cols 48..63 unused)
  gemm_f16_wmma<<<dim3(1, MROWS / 64, 3), 128, 0, stream>>>(
      xc16, (long)MROWS * 256, xpw16, (long)64 * 256, dbc, (long)MROWS * DBC_LD,
      MROWS, 64, 256);

  pack_dtr<<<grid1(3 * MROWS * 32), 256, 0, stream>>>(dbc, dtr16);

  // dt: dtlin[m] = dt_r @ dt_w[m]^T   (K padded 16->32)
  gemm_f16_wmma<<<dim3(256 / 64, MROWS / 64, 3), 128, 0, stream>>>(
      dtr16, (long)MROWS * 32, dtw16, (long)256 * 32, dtlin, (long)MROWS * 256,
      MROWS, 256, 32);

  ssm_scan<<<dim3(3 * BN), 256, 0, stream>>>(dbc, dtlin, dt_b, A_log, D_skip,
                                             xc32, xz, yg16);

  // out_proj: outs[m] = y_gated @ out_w[m]^T
  gemm_f16_wmma<<<dim3(256 / 64, MROWS / 64, 3), 128, 0, stream>>>(
      yg16, (long)MROWS * 256, outw16, (long)256 * 256, outs, (long)MROWS * 256,
      MROWS, 256, 256);

  act_zrh<<<grid1(3 * MROWS * 256), 256, 0, stream>>>(outs);

  // ---- interactive bidirectional GRU (sequential over p; fwd/bwd of one step
  //      never touch the same time index, so each step is one race-free launch set)
  hipMemsetAsync(hf, 0, sizeof(float) * BN * 32 * 256, stream);
  hipMemsetAsync(hb, 0, sizeof(float) * BN * 32 * 256, stream);

  for (int p = 0; p < NPATCH; ++p) {
    if (p > 0) {
      pack_gru<<<grid1(2 * BN * 512), 256, 0, stream>>>(hf, hb, Acat16, p);
      // hif/hib = Acat @ {wf,wb}^T   (M=256, N=256, K=512, batch=2)
      gemm_f16_wmma<<<dim3(256 / 64, 256 / 64, 2), 128, 0, stream>>>(
          Acat16, (long)256 * 512, wcat16, (long)256 * 512, gtmp, (long)256 * 256,
          256, 256, 512);
    }
    gru_update<<<grid1(2 * BN * 256), 256, 0, stream>>>(gtmp, outs, bf, bb, hf, hb,
                                                        p, p > 0 ? 1 : 0);
  }

  final_ln<<<dim3(BN), 256, 0, stream>>>(hf, hb, ln_g, ln_b, out);
  (void)in_sizes; (void)n_in; (void)out_size; (void)ws_size;
}
